// self_LSTM_sparse_attn_764504178979
// MI455X (gfx1250) — matrix operand order repaired and checked by reading, not compiled, not run on hardware
//
#include <hip/hip_runtime.h>
#include <hip/hip_bf16.h>
#include <math.h>

typedef __attribute__((ext_vector_type(16))) _Float16 v16h;
typedef __attribute__((ext_vector_type(8)))  _Float16 v8h;
typedef __attribute__((ext_vector_type(8)))  float    v8f;

#define T_DIM 256
#define B_DIM 64
#define I_DIM 256
#define H_DIM 512
#define G_DIM 2048   // 4*H
#define C_DIM 64
#define TOPK  5
#define EPS_  1e-7f
#define FLT_BIG 3.402823466e38f

union FragU { v16h v; v8h h[2]; };

// A-matrix 16x32 f16 fragment (ISA 7.12.2): lanes 0-15 row M=lane hold K {0..7,16..23},
// lanes 16-31 hold K {8..15,24..31}. Source row-major, stride in halves (mult of 8).
__device__ __forceinline__ v16h load_fragA_h(const _Float16* base, int row0, int stride, int k0) {
  int lane = threadIdx.x & 31;
  const _Float16* p = base + (size_t)(row0 + (lane & 15)) * stride + k0 + ((lane >> 4) << 3);
  FragU f;
  f.h[0] = *(const v8h*)p;
  f.h[1] = *(const v8h*)(p + 16);
  return f.v;
}

// Same A fragment but converting from f32 source on the fly.
__device__ __forceinline__ v16h load_fragA_f32(const float* base, int row0, int stride, int k0) {
  int lane = threadIdx.x & 31;
  const float* p = base + (size_t)(row0 + (lane & 15)) * stride + k0 + ((lane >> 4) << 3);
  v8f a = *(const v8f*)p;
  v8f b = *(const v8f*)(p + 16);
  v8h h0, h1;
#pragma unroll
  for (int j = 0; j < 8; ++j) { h0[j] = (_Float16)a[j]; h1[j] = (_Float16)b[j]; }
  FragU f; f.h[0] = h0; f.h[1] = h1;
  return f.v;
}

// B-matrix 32x16 f16 fragment: lane = column N; lanes 0-15 hold K {0..7,16..23},
// lanes 16-31 hold K {8..15,24..31}. We compute C=A*W^T, so B column n == W row n.
__device__ __forceinline__ v16h load_fragB_h(const _Float16* base, int row0, int stride, int k0) {
  int lane = threadIdx.x & 31;
  const _Float16* p = base + (size_t)(row0 + (lane & 15)) * stride + k0 + ((lane >> 4) << 3);
  FragU f;
  f.h[0] = *(const v8h*)p;
  f.h[1] = *(const v8h*)(p + 16);
  return f.v;
}

__device__ __forceinline__ float sigmoidf_(float x) { return 1.0f / (1.0f + __expf(-x)); }

// Async DMA 16B global->LDS per lane (GLOBAL_LOAD_ASYNC_TO_LDS_B128, tracked by ASYNCcnt).
__device__ __forceinline__ void async_copy16(unsigned lds_addr, const _Float16* g) {
  asm volatile("global_load_async_to_lds_b128 %0, %1, off"
               :: "v"(lds_addr), "v"(g)
               : "memory");
}
__device__ __forceinline__ void wait_async0() {
  asm volatile("s_wait_asynccnt 0x0" ::: "memory");
}

// ---------------- prep: f32->f16 conversions, bias fold, state zeroing ----------------
__global__ __launch_bounds__(256) void prep_kernel(const float* x, const float* W_ih, const float* W_hh,
                            const float* b_ih, const float* b_hh, const float* fc_w,
                            _Float16* x16, _Float16* wih16, _Float16* whh16, _Float16* fcw16,
                            float* b4, float* cstate, _Float16* h16a, float* S)
{
  const int NX  = T_DIM*B_DIM*I_DIM;
  const int NWI = G_DIM*I_DIM;
  const int NWH = G_DIM*H_DIM;
  const int NFC = C_DIM*2*H_DIM;
  const int NB4 = G_DIM;
  const int NC  = B_DIM*H_DIM;
  const int NS  = B_DIM*T_DIM;
  const int total = NX+NWI+NWH+NFC+NB4+NC+NC+NS;
  for (int idx = blockIdx.x*blockDim.x + threadIdx.x; idx < total; idx += gridDim.x*blockDim.x) {
    int j = idx;
    if (j < NX)  { x16[j]   = (_Float16)x[j];    continue; } j -= NX;
    if (j < NWI) { wih16[j] = (_Float16)W_ih[j]; continue; } j -= NWI;
    if (j < NWH) { whh16[j] = (_Float16)W_hh[j]; continue; } j -= NWH;
    if (j < NFC) { fcw16[j] = (_Float16)fc_w[j]; continue; } j -= NFC;
    if (j < NB4) { b4[j] = b_ih[j] + b_hh[j];    continue; } j -= NB4;
    if (j < NC)  { cstate[j] = 0.f;              continue; } j -= NC;
    if (j < NC)  { h16a[j] = (_Float16)0.f;      continue; } j -= NC;
    S[j] = 0.f;
  }
}

// ---------------- Gx = X @ W_ih^T + (b_ih+b_hh): (16384 x 2048, K=256) ----------------
__global__ __launch_bounds__(256) void gx_gemm_kernel(const _Float16* x16, const _Float16* wih16,
                                                      const float* b4, _Float16* Gx)
{
  const int wave = threadIdx.x >> 5;
  const int lane = threadIdx.x & 31;
  int tile = blockIdx.x * 8 + wave;     // 131072 tiles = 1024 M x 128 N
  int nt = tile & 127;
  int mt = tile >> 7;
  int m0 = mt << 4, n0 = nt << 4;
  v8f acc = {};
#pragma unroll
  for (int kk = 0; kk < 8; ++kk) {
    v16h a = load_fragA_h(x16,   m0, I_DIM, kk*32);
    v16h b = load_fragB_h(wih16, n0, I_DIM, kk*32);
    acc = __builtin_amdgcn_wmma_f32_16x16x32_f16(false, a, false, b, (short)0, acc, false, false);
  }
  int n  = n0 + (lane & 15);
  int mo = (lane >> 4) << 3;
  float bias = b4[n];
#pragma unroll
  for (int v = 0; v < 8; ++v) {
    int m = m0 + mo + v;
    Gx[(size_t)m * G_DIM + n] = (_Float16)(acc[v] + bias);
  }
}

// ---------------- per-step attention weights (single block) ----------------
// S[b,t] = tanh(mem[t,b,:]) . wt[H:2H] cached incrementally; batch-constant part of
// the reference score cancels in (score - delta), so it is omitted exactly.
__global__ __launch_bounds__(256) void attn_kernel(const float* h_seq, const float* w_t, float* S,
                                                   int* nz, int* widx, float* wval, int step)
{
  __shared__ float red[256];
  int tid = threadIdx.x;
  if (step > 0) {   // S[b, step] from h_out of step-1 (== mem[step])
    int b = tid & 63, part = tid >> 6;
    const float* hp = h_seq + ((size_t)(step-1)*B_DIM + b) * H_DIM;
    const float* wp = w_t + H_DIM;
    float s = 0.f;
    for (int d = part*128; d < part*128 + 128; ++d) s += tanhf(hp[d]) * wp[d];
    red[tid] = s;
    __syncthreads();
    if (part == 0) S[b*T_DIM + step] = red[b] + red[b+64] + red[b+128] + red[b+192];
    __syncthreads();
  }
  if (tid < 64) {
    const float* Sb = S + tid*T_DIM;
    float top[TOPK];
#pragma unroll
    for (int j = 0; j < TOPK; ++j) top[j] = -FLT_BIG;
    float mn = FLT_BIG;
    int nvalid = step + 1;
    for (int t = 0; t < nvalid; ++t) {
      float s = Sb[t];
      mn = fminf(mn, s);
      if (s > top[TOPK-1]) {
        top[TOPK-1] = s;
#pragma unroll
        for (int j = TOPK-1; j > 0; --j)
          if (top[j] > top[j-1]) { float tmp = top[j]; top[j] = top[j-1]; top[j-1] = tmp; }
      }
    }
    float delta = ((nvalid <= TOPK) ? mn : top[TOPK-1]) + EPS_;
    float sum = 0.f;
    for (int t = 0; t < nvalid; ++t) sum += fmaxf(Sb[t] - delta, 0.f);
    float inv = 1.0f / (sum + EPS_);
    int c = 0;
    for (int t = 1; t < nvalid; ++t) {   // t=0 row of mem is zero -> skip (weight still in sum)
      float w = fmaxf(Sb[t] - delta, 0.f);
      if (w > 0.f) { widx[tid*T_DIM + c] = t; wval[tid*T_DIM + c] = w * inv; ++c; }
    }
    nz[tid] = c;
  }
}

// ---------------- per-step: gates GEMM (64x2048,K=512) + cell + sparse m_t ----------------
// h_prev (64x512 f16, 64KB) is DMA'd global->LDS once (ASYNCcnt), then WMMA A-fragments
// come from LDS (ds_load_b128); W_hh stays in L2 and feeds B-fragments directly.
__global__ __launch_bounds__(256) void step_kernel(const _Float16* whh16, const _Float16* Gx,
                                                   const _Float16* h16_in, _Float16* h16_out,
                                                   float* cstate, float* h_seq, float* m_seq,
                                                   const int* nz, const int* widx, const float* wval,
                                                   int step)
{
  __shared__ _Float16 shA[B_DIM * H_DIM];   // 64 KB staged h_prev
  const int tid  = threadIdx.x;
  const int wave = tid >> 5;
  const int lane = tid & 31;
  const int h0   = blockIdx.x << 6;     // 64 H-columns per block (8 blocks)
  const int mt   = wave & 3;            // 4 M tiles cover B=64
  const int m0   = mt << 4;
  const int half = wave >> 2;           // two waves split the 4 H-subtiles

  // ---- async DMA: 65536 bytes = 4096 x 16B chunks over 256 threads ----
  {
    unsigned lds0 = (unsigned)(unsigned long long)&shA[0];
#pragma unroll
    for (int j = 0; j < 16; ++j) {
      int c = j * 256 + tid;                       // chunk index
      async_copy16(lds0 + c * 16, h16_in + c * 8);
    }
  }
  // prefetch this step's precomputed x-gate rows while DMA is in flight
  __builtin_prefetch(Gx + (size_t)(step*B_DIM + m0) * G_DIM + h0, 0, 1);
  wait_async0();
  __syncthreads();

  v8f acc[2][4] = {};
  for (int kk = 0; kk < 16; ++kk) {
    int k0 = kk << 5;
    // A fragment from LDS
    const _Float16* ap = &shA[(size_t)(m0 + (lane & 15)) * H_DIM + k0 + ((lane >> 4) << 3)];
    FragU af;
    af.h[0] = *(const v8h*)ap;
    af.h[1] = *(const v8h*)(ap + 16);
    v16h a = af.v;
#pragma unroll
    for (int s = 0; s < 2; ++s) {
      int ns = half*2 + s;
#pragma unroll
      for (int g = 0; g < 4; ++g) {
        v16h b = load_fragB_h(whh16, g*H_DIM + h0 + ns*16, H_DIM, k0);
        acc[s][g] = __builtin_amdgcn_wmma_f32_16x16x32_f16(false, a, false, b, (short)0,
                                                           acc[s][g], false, false);
      }
    }
  }
  const int mo   = (lane >> 4) << 3;
  const int ncol = lane & 15;
#pragma unroll
  for (int s = 0; s < 2; ++s) {
    int ns = half*2 + s;
    int hcol = h0 + ns*16 + ncol;
#pragma unroll
    for (int v = 0; v < 8; ++v) {
      int m = m0 + mo + v;                       // batch index
      size_t gxrow = (size_t)(step*B_DIM + m) * G_DIM;
      float gi = acc[s][0][v] + (float)Gx[gxrow + 0*H_DIM + hcol];
      float gf = acc[s][1][v] + (float)Gx[gxrow + 1*H_DIM + hcol];
      float gg = acc[s][2][v] + (float)Gx[gxrow + 2*H_DIM + hcol];
      float go = acc[s][3][v] + (float)Gx[gxrow + 3*H_DIM + hcol];
      size_t ci = (size_t)m*H_DIM + hcol;
      float cp = cstate[ci];
      float cn = sigmoidf_(gf)*cp + sigmoidf_(gi)*tanhf(gg);
      float hn = sigmoidf_(go)*tanhf(cn);
      cstate[ci] = cn;
      float mv = 0.f;                            // sparse attention readout
      int cnt = nz[m];
      for (int j = 0; j < cnt; ++j) {
        int t = widx[m*T_DIM + j];               // mem[t] == h_seq[t-1]
        mv += wval[m*T_DIM + j] * h_seq[((size_t)(t-1)*B_DIM + m)*H_DIM + hcol];
      }
      float ho = hn + mv;
      size_t oi = ((size_t)step*B_DIM + m)*H_DIM + hcol;
      h_seq[oi] = ho;
      m_seq[oi] = mv;
      h16_out[ci] = (_Float16)ho;                // next step's GEMM A operand
    }
  }
}

// ---------------- final FC: (16384 x 64, K=1024 over [h_seq|m_seq]) ----------------
__global__ __launch_bounds__(256) void fc_kernel(const float* h_seq, const float* m_seq,
                                                 const _Float16* fcw16, const float* fc_b, float* out)
{
  const int wave = threadIdx.x >> 5;
  const int lane = threadIdx.x & 31;
  int tile = blockIdx.x * 8 + wave;     // 4096 tiles = 1024 M x 4 N
  int nt = tile & 3;
  int mt = tile >> 2;
  int m0 = mt << 4, n0 = nt << 4;
  v8f acc = {};
  for (int kk = 0; kk < 32; ++kk) {
    int k0 = kk << 5;
    const float* abase = (k0 < H_DIM) ? h_seq : m_seq;
    int ka = (k0 < H_DIM) ? k0 : (k0 - H_DIM);
    v16h a = load_fragA_f32(abase, m0, H_DIM, ka);
    v16h b = load_fragB_h(fcw16, n0, 2*H_DIM, k0);
    acc = __builtin_amdgcn_wmma_f32_16x16x32_f16(false, a, false, b, (short)0, acc, false, false);
  }
  int n  = n0 + (lane & 15);
  int mo = (lane >> 4) << 3;
  float bias = fc_b[n];
#pragma unroll
  for (int v = 0; v < 8; ++v) {
    int m = m0 + mo + v;
    out[(size_t)m*C_DIM + n] = acc[v] + bias;
  }
}

extern "C" void kernel_launch(void* const* d_in, const int* in_sizes, int n_in,
                              void* d_out, int out_size, void* d_ws, size_t ws_size,
                              hipStream_t stream)
{
  (void)in_sizes; (void)n_in; (void)out_size; (void)ws_size;
  const float* x    = (const float*)d_in[0];
  const float* W_ih = (const float*)d_in[1];
  const float* W_hh = (const float*)d_in[2];
  const float* b_ih = (const float*)d_in[3];
  const float* b_hh = (const float*)d_in[4];
  const float* w_t  = (const float*)d_in[5];   // (2H,1) flat
  const float* fc_w = (const float*)d_in[6];
  const float* fc_b = (const float*)d_in[7];
  float* out = (float*)d_out;

  char* ws = (char*)d_ws;
  size_t off = 0;
  auto alloc = [&](size_t bytes) -> void* {
    void* p = ws + off; off += (bytes + 255) & ~(size_t)255; return p;
  };
  float*    h_seq  = (float*)   alloc((size_t)T_DIM*B_DIM*H_DIM*4);   // h_out history (== mem[t+1])
  float*    m_seq  = (float*)   alloc((size_t)T_DIM*B_DIM*H_DIM*4);
  _Float16* Gx     = (_Float16*)alloc((size_t)T_DIM*B_DIM*G_DIM*2);   // precomputed x-gates
  _Float16* x16    = (_Float16*)alloc((size_t)T_DIM*B_DIM*I_DIM*2);
  _Float16* wih16  = (_Float16*)alloc((size_t)G_DIM*I_DIM*2);
  _Float16* whh16  = (_Float16*)alloc((size_t)G_DIM*H_DIM*2);
  _Float16* fcw16  = (_Float16*)alloc((size_t)C_DIM*2*H_DIM*2);
  float*    b4     = (float*)   alloc((size_t)G_DIM*4);
  float*    cstate = (float*)   alloc((size_t)B_DIM*H_DIM*4);
  _Float16* h16a   = (_Float16*)alloc((size_t)B_DIM*H_DIM*2);         // double-buffered h (f16)
  _Float16* h16b   = (_Float16*)alloc((size_t)B_DIM*H_DIM*2);
  float*    S      = (float*)   alloc((size_t)B_DIM*T_DIM*4);         // cached memory-side scores
  int*      nzc    = (int*)     alloc((size_t)B_DIM*4);
  int*      widx   = (int*)     alloc((size_t)B_DIM*T_DIM*4);
  float*    wval   = (float*)   alloc((size_t)B_DIM*T_DIM*4);

  prep_kernel<<<4096, 256, 0, stream>>>(x, W_ih, W_hh, b_ih, b_hh, fc_w,
                                        x16, wih16, whh16, fcw16, b4, cstate, h16a, S);
  gx_gemm_kernel<<<16384, 256, 0, stream>>>(x16, wih16, b4, Gx);
  for (int i = 0; i < T_DIM; ++i) {
    attn_kernel<<<1, 256, 0, stream>>>(h_seq, w_t, S, nzc, widx, wval, i);
    const _Float16* hin = (i & 1) ? h16b : h16a;
    _Float16*      hout = (i & 1) ? h16a : h16b;
    step_kernel<<<8, 256, 0, stream>>>(whh16, Gx, hin, hout, cstate, h_seq, m_seq,
                                       nzc, widx, wval, i);
  }
  fc_kernel<<<512, 256, 0, stream>>>(h_seq, m_seq, fcw16, fc_b, out);
}
